// SelfAttention_65481071399904
// MI455X (gfx1250) — matrix operand order repaired and checked by reading, not compiled, not run on hardware
//
#include <hip/hip_runtime.h>
#include <hip/hip_bf16.h>

#define EMBED   1024
#define NQKV    3072
#define HEADS   16
#define HDIM    64
#define TOK_PER_BLK 32

#define AS1 __attribute__((address_space(1)))
#define AS3 __attribute__((address_space(3)))

typedef __attribute__((ext_vector_type(16))) __bf16 v16bf;
typedef __attribute__((ext_vector_type(8)))  __bf16 v8bf;
typedef __attribute__((ext_vector_type(4)))  __bf16 v4bf;
typedef __attribute__((ext_vector_type(8)))  float  v8f;
typedef __attribute__((ext_vector_type(4)))  float  v4f;
typedef __attribute__((ext_vector_type(4)))  int    v4i;

union ABfrag { v16bf v; v8bf h[2]; };

#if defined(__has_builtin)
#  if __has_builtin(__builtin_amdgcn_global_load_async_to_lds_b128)
#    define HAVE_ASYNC_LDS 1
#  endif
#endif
#ifndef HAVE_ASYNC_LDS
#  define HAVE_ASYNC_LDS 0
#endif

__device__ __forceinline__ void wait_async0() {
#if defined(__has_builtin) && __has_builtin(__builtin_amdgcn_s_wait_asynccnt)
    __builtin_amdgcn_s_wait_asynccnt(0);
#else
    asm volatile("s_wait_asynccnt 0x0" ::: "memory");
#endif
}

// Native bf16 convert: backend emits gfx1250 HW cvt (packed where possible).
__device__ __forceinline__ __bf16 f2bf(float f) { return (__bf16)f; }

__device__ __forceinline__ v8f wmma_bf16(v16bf a, v16bf b, v8f c) {
    // D = A(16x32 bf16) * B(32x16 bf16) + C(16x16 f32)
    return __builtin_amdgcn_wmma_f32_16x16x32_bf16(
        false, a, false, b, (short)0, c, false, false);
}

__device__ __forceinline__ v8f v8f_zero() {
    v8f z;
#pragma unroll
    for (int i = 0; i < 8; ++i) z[i] = 0.0f;
    return z;
}

__device__ __forceinline__ v8bf v8bf_zero() {
    v8bf z;
#pragma unroll
    for (int i = 0; i < 8; ++i) z[i] = (__bf16)0.0f;
    return z;
}

// Stage a [rows x 1024] bf16 tile from global into padded LDS rows using the
// CDNA5 async global->LDS path (ASYNCcnt).
__device__ __forceinline__ void stage_tile(const __bf16* __restrict__ src,
                                           __bf16 (*dst)[EMBED + 8],
                                           int rows, int tid) {
#pragma unroll 1
    for (int c = tid; c < rows * 128; c += 256) {        // 16B chunks
        const int r  = c >> 7;
        const int c8 = (c & 127) * 8;
#if HAVE_ASYNC_LDS
        __builtin_amdgcn_global_load_async_to_lds_b128(
            (AS1 v4i*)(src + (size_t)r * EMBED + c8),
            (AS3 v4i*)&dst[r][c8], 0, 0);
#else
        *(v8bf*)&dst[r][c8] = *(const v8bf*)(src + (size_t)r * EMBED + c8);
#endif
    }
#if HAVE_ASYNC_LDS
    wait_async0();
#endif
}

// ---------------------------------------------------------------------------
// Kernel: convert f32 weight [rows][cols] -> bf16 transposed [cols][rows]
// ---------------------------------------------------------------------------
__global__ __launch_bounds__(256) void k_transpose_bf16(
        const float* __restrict__ src, __bf16* __restrict__ dst,
        int rows, int cols) {
    __shared__ float tile[32][33];
    const int t  = threadIdx.x;
    const int lx = t & 31;
    const int ly = t >> 5;
    const int c0 = blockIdx.x * 32;
    const int r0 = blockIdx.y * 32;
#pragma unroll
    for (int i = 0; i < 32; i += 8)
        tile[ly + i][lx] = src[(size_t)(r0 + ly + i) * cols + (c0 + lx)];
    __syncthreads();
#pragma unroll
    for (int i = 0; i < 32; i += 8)
        dst[(size_t)(c0 + ly + i) * rows + (r0 + lx)] = f2bf(tile[lx][ly + i]);
}

// ---------------------------------------------------------------------------
// Kernel: elementwise f32 -> bf16 (for x), 4 elements/thread
// ---------------------------------------------------------------------------
__global__ __launch_bounds__(256) void k_f32_to_bf16(
        const float* __restrict__ src, __bf16* __restrict__ dst) {
    const size_t i = ((size_t)blockIdx.x * 256 + threadIdx.x) * 4;
    const v4f v = *(const v4f*)(src + i);
    v4bf o;
#pragma unroll
    for (int j = 0; j < 4; ++j) o[j] = f2bf(v[j]);
    *(v4bf*)(dst + i) = o;
}

// ---------------------------------------------------------------------------
// Kernel 1: fused QKV projection + per-token head-attention.
//   Block = 256 threads (8 waves), owns 32 tokens (2 M-tiles).
//   Phase A: qkv[32 x 3072] = x_tile[32 x 1024] @ W_qkv  (bf16 WMMA -> LDS)
//   Phase B: per token: S = Q K^T * 0.125 ; P = softmax(S) ; O = P V (WMMA)
// ---------------------------------------------------------------------------
__global__ __launch_bounds__(256) void k_qkv_attn(
        const __bf16* __restrict__ xb,     // [32768][1024] bf16
        const __bf16* __restrict__ Wt,     // [3072][1024]  bf16 (transposed)
        __bf16*       __restrict__ o) {    // [32768][1024] bf16
    __shared__ __bf16 sA[TOK_PER_BLK][EMBED + 8];   // x tile (66 KB)
    __shared__ __bf16 sQKV[TOK_PER_BLK][NQKV];      // qkv tile (192 KB)
    __shared__ __bf16 sP[8][16][16];                // per-wave probs (4 KB)

    const int tid  = threadIdx.x;
    const int wave = tid >> 5;
    const int lane = tid & 31;
    const int l15  = lane & 15;
    const int half = lane >> 4;
    const size_t tok0 = (size_t)blockIdx.x * TOK_PER_BLK;

    stage_tile(xb + tok0 * EMBED, sA, TOK_PER_BLK, tid);
    __syncthreads();

    // ---- Phase A: 192 N-tiles x 2 M-tiles; wave w -> nt = w + 8i, 6 groups of 4
#pragma unroll 1
    for (int g = 0; g < 6; ++g) {
        v8f acc[8];
#pragma unroll
        for (int i = 0; i < 8; ++i) acc[i] = v8f_zero();

#pragma unroll 1
        for (int k0 = 0; k0 < EMBED; k0 += 32) {
            ABfrag a0, a1;   // A 16x32: lanes0-15 K=0..15, lanes16-31 K=16..31
            a0.h[0] = *(const v8bf*)&sA[l15][k0 + half * 16];
            a0.h[1] = *(const v8bf*)&sA[l15][k0 + half * 16 + 8];
            a1.h[0] = *(const v8bf*)&sA[16 + l15][k0 + half * 16];
            a1.h[1] = *(const v8bf*)&sA[16 + l15][k0 + half * 16 + 8];
#pragma unroll
            for (int ii = 0; ii < 4; ++ii) {
                const int col = (wave + 8 * (g * 4 + ii)) * 16 + l15;
                // B 32x16: lanes0-15 K=k0..k0+15, lanes16-31 K=k0+16..k0+31
                const v16bf b = *(const v16bf*)&Wt[(size_t)col * EMBED + k0 + half * 16];
                acc[ii * 2]     = wmma_bf16(a0.v, b, acc[ii * 2]);
                acc[ii * 2 + 1] = wmma_bf16(a1.v, b, acc[ii * 2 + 1]);
            }
        }
#pragma unroll
        for (int ii = 0; ii < 4; ++ii) {
            const int n = (wave + 8 * (g * 4 + ii)) * 16 + l15;
#pragma unroll
            for (int mt = 0; mt < 2; ++mt)
#pragma unroll
                for (int r = 0; r < 8; ++r)
                    sQKV[mt * 16 + r + half * 8][n] = f2bf(acc[ii * 2 + mt][r]);
        }
    }
    __syncthreads();

    // ---- Phase B: attention; wave handles tokens wave + 8*tt, tt = 0..3 ----
    const v8bf z8 = v8bf_zero();
#pragma unroll 1
    for (int tt = 0; tt < 4; ++tt) {
        const int t = wave + tt * 8;

        // S[h][g] = sum_d Q[h][d] K[g][d]
        v8f s = v8f_zero();
#pragma unroll
        for (int k0 = 0; k0 < HDIM; k0 += 32) {
            ABfrag a;
            a.h[0] = *(const v8bf*)&sQKV[t][l15 * 192 + k0 + half * 16];
            a.h[1] = *(const v8bf*)&sQKV[t][l15 * 192 + k0 + half * 16 + 8];
            const v16bf b = *(const v16bf*)&sQKV[t][l15 * 192 + 64 + k0 + half * 16];
            s = wmma_bf16(a.v, b, s);
        }

        // softmax over g (N dim = 16 lanes of a half), per C-row VGPR
        float p[8];
#pragma unroll
        for (int r = 0; r < 8; ++r) {
            float v = s[r] * 0.125f;           // D^-0.5
            float mx = v;
#pragma unroll
            for (int off = 1; off < 16; off <<= 1)
                mx = fmaxf(mx, __shfl_xor(mx, off, 32));
            const float e = __expf(v - mx);
            float sm = e;
#pragma unroll
            for (int off = 1; off < 16; off <<= 1)
                sm += __shfl_xor(sm, off, 32);
            p[r] = e / sm;
        }

#pragma unroll
        for (int r = 0; r < 8; ++r)
            sP[wave][r + half * 8][l15] = f2bf(p[r]);
        __syncthreads();   // uniform trip-count across waves

        // A = P padded 16x16 -> 16x32 (upper K zero)
        ABfrag pa;
        pa.h[0] = half ? z8 : *(const v8bf*)&sP[wave][l15][0];
        pa.h[1] = half ? z8 : *(const v8bf*)&sP[wave][l15][8];

        // O[h][d] = sum_g P[h][g] V[g][d]
#pragma unroll
        for (int dt = 0; dt < 4; ++dt) {
            v16bf b;
            if (half == 0) {   // K=0..15 hold V rows; K=16..31 zero (NaN prop.)
#pragma unroll
                for (int j = 0; j < 8; ++j) {
                    b[2 * j]     = sQKV[t][(2 * j) * 192 + 128 + dt * 16 + l15];
                    b[2 * j + 1] = sQKV[t][(2 * j + 1) * 192 + 128 + dt * 16 + l15];
                }
            } else {
#pragma unroll
                for (int j = 0; j < 16; ++j) b[j] = (__bf16)0.0f;
            }
            const v8f av = wmma_bf16(pa.v, b, v8f_zero());
#pragma unroll
            for (int r = 0; r < 8; ++r)
                o[(tok0 + t) * EMBED + (size_t)(r + half * 8) * HDIM + dt * 16 + l15]
                    = f2bf(av[r]);
        }
        __syncthreads();
    }
}

// ---------------------------------------------------------------------------
// Kernel 2: out = o @ W_out + b_out  (bf16 WMMA, f32 out). 32 tokens/block.
// ---------------------------------------------------------------------------
__global__ __launch_bounds__(256) void k_out_proj(
        const __bf16* __restrict__ oin,   // [32768][1024] bf16
        const __bf16* __restrict__ Wt,    // [1024][1024]  bf16 (transposed)
        const float*  __restrict__ bias,  // [1024]
        float*        __restrict__ out) { // [32768][1024] f32
    __shared__ __bf16 sA[TOK_PER_BLK][EMBED + 8];

    const int tid  = threadIdx.x;
    const int wave = tid >> 5;
    const int lane = tid & 31;
    const int l15  = lane & 15;
    const int half = lane >> 4;
    const size_t tok0 = (size_t)blockIdx.x * TOK_PER_BLK;

    stage_tile(oin + tok0 * EMBED, sA, TOK_PER_BLK, tid);
    __syncthreads();

    // 64 N-tiles x 2 M-tiles; wave w -> nt = w*8 + (g*4+ii)
#pragma unroll 1
    for (int g = 0; g < 2; ++g) {
        v8f acc[8];
#pragma unroll
        for (int i = 0; i < 8; ++i) acc[i] = v8f_zero();

#pragma unroll 1
        for (int k0 = 0; k0 < EMBED; k0 += 32) {
            ABfrag a0, a1;
            a0.h[0] = *(const v8bf*)&sA[l15][k0 + half * 16];
            a0.h[1] = *(const v8bf*)&sA[l15][k0 + half * 16 + 8];
            a1.h[0] = *(const v8bf*)&sA[16 + l15][k0 + half * 16];
            a1.h[1] = *(const v8bf*)&sA[16 + l15][k0 + half * 16 + 8];
#pragma unroll
            for (int ii = 0; ii < 4; ++ii) {
                const int col = (wave * 8 + g * 4 + ii) * 16 + l15;
                const v16bf b = *(const v16bf*)&Wt[(size_t)col * EMBED + k0 + half * 16];
                acc[ii * 2]     = wmma_bf16(a0.v, b, acc[ii * 2]);
                acc[ii * 2 + 1] = wmma_bf16(a1.v, b, acc[ii * 2 + 1]);
            }
        }
#pragma unroll
        for (int ii = 0; ii < 4; ++ii) {
            const int col = (wave * 8 + g * 4 + ii) * 16 + l15;
            const float bv = bias[col];
#pragma unroll
            for (int mt = 0; mt < 2; ++mt)
#pragma unroll
                for (int r = 0; r < 8; ++r)
                    out[(tok0 + mt * 16 + r + half * 8) * EMBED + col]
                        = acc[ii * 2 + mt][r] + bv;
        }
    }
}

// ---------------------------------------------------------------------------
extern "C" void kernel_launch(void* const* d_in, const int* in_sizes, int n_in,
                              void* d_out, int out_size, void* d_ws, size_t ws_size,
                              hipStream_t stream) {
    (void)in_sizes; (void)n_in; (void)out_size; (void)ws_size;
    const float* x     = (const float*)d_in[0];   // [8][4096][1024]
    const float* W_qkv = (const float*)d_in[1];   // [1024][3072]
    const float* W_out = (const float*)d_in[2];   // [1024][1024]
    const float* b_out = (const float*)d_in[3];   // [1024]
    float* out = (float*)d_out;

    char* ws = (char*)d_ws;
    __bf16* Wt_qkv = (__bf16*)(ws);                          //  6 MB
    __bf16* Wt_out = (__bf16*)(ws + 6291456);                //  2 MB
    __bf16* obuf   = (__bf16*)(ws + 8388608);                // 64 MB
    __bf16* xb     = (__bf16*)(ws + 8388608 + 67108864);     // 64 MB

    const int tokens = 8 * 4096;                              // 32768

    k_transpose_bf16<<<dim3(NQKV / 32, EMBED / 32), 256, 0, stream>>>(W_qkv, Wt_qkv, EMBED, NQKV);
    k_transpose_bf16<<<dim3(EMBED / 32, EMBED / 32), 256, 0, stream>>>(W_out, Wt_out, EMBED, EMBED);
    k_f32_to_bf16<<<(tokens * (EMBED / 1024)), 256, 0, stream>>>(x, xb);  // 32768 blocks * 1024 elems

    k_qkv_attn<<<tokens / TOK_PER_BLK, 256, 0, stream>>>(xb, Wt_qkv, obuf);
    k_out_proj<<<tokens / TOK_PER_BLK, 256, 0, stream>>>(obuf, Wt_out, b_out, out);
}
